// NestedTensorBlock_87007447482980
// MI455X (gfx1250) — matrix operand order repaired and checked by reading, not compiled, not run on hardware
//
#include <hip/hip_runtime.h>
#include <stdint.h>

// ---------------------------------------------------------------------------
// MI455X (gfx1250) ViT block: LN -> QKV -> FMHA -> proj(+LS residual)
//                             -> LN -> fc1+GELU -> fc2(+LS residual)
// All matmuls on v_wmma_f32_16x16x32_bf16 (wave32). GEMM staging uses
// GLOBAL_LOAD_ASYNC_TO_LDS_B128 + s_wait_asynccnt double-buffering when the
// toolchain exposes the builtins (guarded; falls back to reg-overlap copies).
// ---------------------------------------------------------------------------

typedef __attribute__((ext_vector_type(16))) __bf16 v16bf;
typedef __attribute__((ext_vector_type(8)))  float  v8f;
typedef int v4i __attribute__((vector_size(16)));   // matches builtin proto

union FragAB { v16bf v; unsigned int u[8]; unsigned short s[16]; };

#if __has_builtin(__builtin_amdgcn_global_load_async_to_lds_b128) && \
    __has_builtin(__builtin_amdgcn_s_wait_asynccnt)
#define HAVE_ASYNC_LDS 1
__device__ __forceinline__ void async_cp16B(const unsigned short* g,
                                            unsigned short* l) {
    __builtin_amdgcn_global_load_async_to_lds_b128(
        (__attribute__((address_space(1))) v4i*)g,
        (__attribute__((address_space(3))) v4i*)l, 0, 0);
}
#else
#define HAVE_ASYNC_LDS 0
#endif

__device__ __forceinline__ unsigned short f2bf(float f) {
    unsigned int u = __float_as_uint(f);
    u += 0x7fffu + ((u >> 16) & 1u);          // round-to-nearest-even
    return (unsigned short)(u >> 16);
}
__device__ __forceinline__ unsigned int pack_bf2(float lo, float hi) {
    return (unsigned int)f2bf(lo) | ((unsigned int)f2bf(hi) << 16);
}

// K-offset patterns from CDNA5 ISA §7.12.2 (16-bit operands).
// A 16x32: lanes 0-15 hold K 0-7 / 16-23, lanes 16-31 hold K 8-15 / 24-31.
__device__ __forceinline__ int koffA(int j, int hl) {
    return (j < 4 ? 2 * j : 8 + 2 * j) + hl * 8;
}
// B 32x16: same K order as A: lanes 0-15 hold K 0-7 / 16-23, lanes 16-31 hold K 8-15 / 24-31.
__device__ __forceinline__ int koffB(int j, int hl) {
    return (j < 4 ? 2 * j : 8 + 2 * j) + hl * 8;
}

// ---------------------------------------------------------------------------
// Weight convert + transpose: W[K,N] f32 -> Wt[N,K] bf16  (once per launch)
// ---------------------------------------------------------------------------
__global__ __launch_bounds__(256)
void wconv_t(const float* __restrict__ W, unsigned short* __restrict__ Wt,
             int K, int N) {
    unsigned int e = blockIdx.x * 256u + threadIdx.x;
    unsigned int k = e / (unsigned int)N;
    unsigned int n = e % (unsigned int)N;
    Wt[(size_t)n * K + k] = f2bf(W[e]);
}

// ---------------------------------------------------------------------------
// LayerNorm over D=1024, output bf16.  One workgroup (256 thr) per token.
// ---------------------------------------------------------------------------
__global__ __launch_bounds__(256)
void ln_bf16(const float* __restrict__ x, const float* __restrict__ w,
             const float* __restrict__ b, unsigned short* __restrict__ o) {
    const int row = blockIdx.x, t = threadIdx.x;
    const float4 f = ((const float4*)(x + (size_t)row * 1024))[t];
    float s = f.x + f.y + f.z + f.w;
    float q = f.x * f.x + f.y * f.y + f.z * f.z + f.w * f.w;
#pragma unroll
    for (int off = 16; off; off >>= 1) {
        s += __shfl_xor(s, off, 32);
        q += __shfl_xor(q, off, 32);
    }
    __shared__ float rs[8], rq[8];
    if ((t & 31) == 0) { rs[t >> 5] = s; rq[t >> 5] = q; }
    __syncthreads();
    s = 0.f; q = 0.f;
#pragma unroll
    for (int i = 0; i < 8; ++i) { s += rs[i]; q += rq[i]; }
    const float mean = s * (1.0f / 1024.0f);
    const float var  = q * (1.0f / 1024.0f) - mean * mean;
    const float rstd = rsqrtf(var + 1e-5f);
    const float4 wv = ((const float4*)w)[t];
    const float4 bv = ((const float4*)b)[t];
    unsigned short* op = o + (size_t)row * 1024 + t * 4;
    op[0] = f2bf((f.x - mean) * rstd * wv.x + bv.x);
    op[1] = f2bf((f.y - mean) * rstd * wv.y + bv.y);
    op[2] = f2bf((f.z - mean) * rstd * wv.z + bv.z);
    op[3] = f2bf((f.w - mean) * rstd * wv.w + bv.w);
}

// ---------------------------------------------------------------------------
// bf16 WMMA GEMM:  C[M,N] = A[M,K] @ Bt[N,K]^T  (+ epilogue)
//   EPI 0: +bias                 -> bf16 out
//   EPI 1: +bias, exact GELU     -> bf16 out
//   EPI 2: res + gamma*(.+bias)  -> f32 out
// WG tile 128x128, 8 waves, wave tile 16x128 (8 WMMAs / K-step, A reused 8x).
// Async path: double-buffered LDS fed by global_load_async_to_lds_b128;
// s_wait_asynccnt(4) relies on in-order async completion per wave.
// ---------------------------------------------------------------------------
template <int EPI>
__global__ __launch_bounds__(256)
void gemm_bf16(const unsigned short* __restrict__ A,
               const unsigned short* __restrict__ Bt,
               const float* __restrict__ bias,
               const float* __restrict__ res,
               const float* __restrict__ gamma,
               unsigned short* __restrict__ out_bf,
               float* __restrict__ out_f,
               int M, int N, int K) {
    const int t    = threadIdx.x;
    const int lane = t & 31, wave = t >> 5;
    const int hl   = lane >> 4, ln16 = lane & 15;
    const int tileM = blockIdx.x * 128, tileN = blockIdx.y * 128;

    v8f acc[8];
#pragma unroll
    for (int f = 0; f < 8; ++f)
#pragma unroll
        for (int v = 0; v < 8; ++v) acc[f][v] = 0.0f;

    // staging role: 128 rows x 32k, 16 shorts per thread
    const int rld = t >> 1, kld = (t & 1) * 16;
    const unsigned short* Ag = A  + (size_t)(tileM + rld) * K + kld;
    const unsigned short* Bg = Bt + (size_t)(tileN + rld) * K + kld;

#if HAVE_ASYNC_LDS
    __shared__ unsigned short As[2][128][40];  // 80B row stride (16B aligned)
    __shared__ unsigned short Bs[2][128][40];

    auto computeTile = [&](int buf) {
        FragAB a;
#pragma unroll
        for (int j = 0; j < 8; ++j)
            a.u[j] = *(const unsigned int*)&As[buf][wave * 16 + ln16][koffA(j, hl)];
#pragma unroll
        for (int f = 0; f < 8; ++f) {
            FragAB bfr;
#pragma unroll
            for (int j = 0; j < 8; ++j)
                bfr.u[j] = *(const unsigned int*)&Bs[buf][f * 16 + ln16][koffB(j, hl)];
            acc[f] = __builtin_amdgcn_wmma_f32_16x16x32_bf16(
                false, a.v, false, bfr.v, (short)0, acc[f], false, false);
        }
    };
    auto issueTile = [&](int k0, int buf) {   // 4 async b128 copies / thread
        async_cp16B(Ag + k0,     &As[buf][rld][kld]);
        async_cp16B(Ag + k0 + 8, &As[buf][rld][kld + 8]);
        async_cp16B(Bg + k0,     &Bs[buf][rld][kld]);
        async_cp16B(Bg + k0 + 8, &Bs[buf][rld][kld + 8]);
    };

    issueTile(0, 0);
    int cur = 0, k0 = 0;
    for (; k0 + 32 < K; k0 += 32) {
        __syncthreads();                       // all waves done reading cur^1
        issueTile(k0 + 32, cur ^ 1);
        __builtin_amdgcn_s_wait_asynccnt(4);   // first 4 (tile cur) landed
        __syncthreads();                       // tile cur visible to all waves
        computeTile(cur);
        cur ^= 1;
    }
    __syncthreads();
    __builtin_amdgcn_s_wait_asynccnt(0);       // last tile landed
    __syncthreads();
    computeTile(cur);
#else
    __shared__ unsigned short As[128][40];
    __shared__ unsigned short Bs[128][40];
    uint4 ra0 = *(const uint4*)(Ag);
    uint4 ra1 = *(const uint4*)(Ag + 8);
    uint4 rb0 = *(const uint4*)(Bg);
    uint4 rb1 = *(const uint4*)(Bg + 8);
    for (int k0 = 0; k0 < K; k0 += 32) {
        __syncthreads();
        *(uint4*)&As[rld][kld]     = ra0;
        *(uint4*)&As[rld][kld + 8] = ra1;
        *(uint4*)&Bs[rld][kld]     = rb0;
        *(uint4*)&Bs[rld][kld + 8] = rb1;
        __syncthreads();
        if (k0 + 32 < K) {                     // overlap next loads w/ compute
            ra0 = *(const uint4*)(Ag + k0 + 32);
            ra1 = *(const uint4*)(Ag + k0 + 40);
            rb0 = *(const uint4*)(Bg + k0 + 32);
            rb1 = *(const uint4*)(Bg + k0 + 40);
            if (k0 + 64 < K) {
                __builtin_prefetch(Ag + k0 + 64, 0, 1);  // global_prefetch_b8
                __builtin_prefetch(Bg + k0 + 64, 0, 1);
            }
        }
        FragAB a;
#pragma unroll
        for (int j = 0; j < 8; ++j)
            a.u[j] = *(const unsigned int*)&As[wave * 16 + ln16][koffA(j, hl)];
#pragma unroll
        for (int f = 0; f < 8; ++f) {
            FragAB bfr;
#pragma unroll
            for (int j = 0; j < 8; ++j)
                bfr.u[j] = *(const unsigned int*)&Bs[f * 16 + ln16][koffB(j, hl)];
            acc[f] = __builtin_amdgcn_wmma_f32_16x16x32_bf16(
                false, a.v, false, bfr.v, (short)0, acc[f], false, false);
        }
    }
#endif

#pragma unroll
    for (int f = 0; f < 8; ++f) {
        const int n  = tileN + f * 16 + ln16;
        const float bi = bias[n];
        const float gm = (EPI == 2) ? gamma[n] : 0.0f;
#pragma unroll
        for (int v = 0; v < 8; ++v) {
            const int r = tileM + wave * 16 + v + hl * 8;
            float xv = acc[f][v] + bi;
            if (EPI == 1)
                xv = 0.5f * xv * (1.0f + erff(xv * 0.70710678118654752f));
            if (EPI == 2)
                out_f[(size_t)r * N + n] = res[(size_t)r * N + n] + gm * xv;
            else
                out_bf[(size_t)r * N + n] = f2bf(xv);
        }
    }
}

// ---------------------------------------------------------------------------
// Flash attention, transposed-score two-pass softmax.
// qkv: [B*S, 3072] bf16 (q|k|v, each [H=16][HD=64]).
// Grid: (B*H, S/64); 128 threads = 4 waves; wave handles 16 query rows.
// Pass 1: m = rowmax via S^T = K@Q^T WMMAs (in-lane max, 1 shfl at end).
// Pass 2: p = exp(s-m) lands directly in the P A-fragment layout (register
//         pack only), l accumulated in-lane, P@V WMMAs vs LDS-staged V^T.
// ---------------------------------------------------------------------------
__global__ __launch_bounds__(128)
void attn_fmha(const unsigned short* __restrict__ qkv,
               unsigned short* __restrict__ o, int S) {
    __shared__ unsigned short Vt[64][34];     // [d][key], shared by 4 waves
    __shared__ float Linv[4][16];             // per-wave 1/l per query row
    const int t    = threadIdx.x;
    const int lane = t & 31, wave = t >> 5;
    const int hl   = lane >> 4, ln16 = lane & 15;
    const int b    = blockIdx.x >> 4, h = blockIdx.x & 15;
    const int q0   = blockIdx.y * 64 + wave * 16;
    const int base = b * S;

    // Q as B-operand of S^T = K @ Q^T: lane = query column, d-pairs koffB.
    FragAB qb[2];
    {
        const unsigned short* qp =
            qkv + (size_t)(base + q0 + ln16) * 3072 + h * 64;
#pragma unroll
        for (int s = 0; s < 2; ++s)
#pragma unroll
            for (int j = 0; j < 8; ++j)
                qb[s].u[j] = *(const unsigned int*)(qp + s * 32 + koffB(j, hl));
    }

    v8f zf;
#pragma unroll
    for (int v = 0; v < 8; ++v) zf[v] = 0.0f;

    const unsigned short* kbase = qkv + (size_t)base * 3072 + 1024 + h * 64;
    const int nkb = S >> 5;

    // ---------------- pass 1: row max (pure VALU reduction) ----------------
    float m = -1.0e30f;
    for (int kb = 0; kb < nkb; ++kb) {
#pragma unroll
        for (int c = 0; c < 2; ++c) {
            const unsigned short* kp =
                kbase + (size_t)(kb * 32 + c * 16 + ln16) * 3072;
            FragAB k0f, k1f;
#pragma unroll
            for (int j = 0; j < 8; ++j) {
                k0f.u[j] = *(const unsigned int*)(kp + koffA(j, hl));
                k1f.u[j] = *(const unsigned int*)(kp + 32 + koffA(j, hl));
            }
            v8f st = __builtin_amdgcn_wmma_f32_16x16x32_bf16(
                false, k0f.v, false, qb[0].v, (short)0, zf, false, false);
            st = __builtin_amdgcn_wmma_f32_16x16x32_bf16(
                false, k1f.v, false, qb[1].v, (short)0, st, false, false);
#pragma unroll
            for (int v = 0; v < 8; ++v) m = fmaxf(m, st[v]);
        }
    }
    m = fmaxf(m, __shfl_xor(m, 16, 32));      // merge key-subset halves
    m *= 0.125f;                               // 1/sqrt(HD), scale > 0

    // ---------------- pass 2: exp, l-sum, P @ V ----------------
    v8f acc[4];
#pragma unroll
    for (int f = 0; f < 4; ++f)
#pragma unroll
        for (int v = 0; v < 8; ++v) acc[f][v] = 0.0f;
    float l = 0.0f;

    const int vkey = t >> 2, vdc = (t & 3) * 16;   // V staging role
    for (int kb = 0; kb < nkb; ++kb) {
        // recompute S^T for this 32-key block (4 WMMAs)
        v8f st0, st1;
#pragma unroll
        for (int c = 0; c < 2; ++c) {
            const unsigned short* kp =
                kbase + (size_t)(kb * 32 + c * 16 + ln16) * 3072;
            FragAB k0f, k1f;
#pragma unroll
            for (int j = 0; j < 8; ++j) {
                k0f.u[j] = *(const unsigned int*)(kp + koffA(j, hl));
                k1f.u[j] = *(const unsigned int*)(kp + 32 + koffA(j, hl));
            }
            v8f st = __builtin_amdgcn_wmma_f32_16x16x32_bf16(
                false, k0f.v, false, qb[0].v, (short)0, zf, false, false);
            st = __builtin_amdgcn_wmma_f32_16x16x32_bf16(
                false, k1f.v, false, qb[1].v, (short)0, st, false, false);
            if (c == 0) st0 = st; else st1 = st;
        }

        // p = exp(score*scale - m); all 16 values belong to query row ln16.
        float p0[8], p1[8];
#pragma unroll
        for (int v = 0; v < 8; ++v) {
            p0[v] = __expf(st0[v] * 0.125f - m);
            p1[v] = __expf(st1[v] * 0.125f - m);
            l += p0[v] + p1[v];
        }
        // S^T C-layout == P A-fragment layout: pack in registers, no LDS.
        FragAB pa;
#pragma unroll
        for (int j = 0; j < 4; ++j) {
            pa.u[j]     = pack_bf2(p0[2 * j], p0[2 * j + 1]);
            pa.u[j + 4] = pack_bf2(p1[2 * j], p1[2 * j + 1]);
        }

        __syncthreads();  // all waves done reading previous Vt
        {   // stage V^T cooperatively: [32 keys][64 d] -> Vt[d][key]
            const unsigned short* vp =
                qkv + (size_t)(base + kb * 32 + vkey) * 3072 + 2048 + h * 64 + vdc;
            uint4 d0 = *(const uint4*)vp;
            uint4 d1 = *(const uint4*)(vp + 8);
            const unsigned int dd[8] = {d0.x, d0.y, d0.z, d0.w,
                                        d1.x, d1.y, d1.z, d1.w};
#pragma unroll
            for (int i = 0; i < 8; ++i) {
                Vt[vdc + 2 * i][vkey]     = (unsigned short)(dd[i] & 0xffffu);
                Vt[vdc + 2 * i + 1][vkey] = (unsigned short)(dd[i] >> 16);
            }
        }
        __syncthreads();

        // acc += P @ V   (B-frags from Vt: lane = d column, key-pairs koffB)
#pragma unroll
        for (int f = 0; f < 4; ++f) {
            FragAB vf;
#pragma unroll
            for (int j = 0; j < 8; ++j)
                vf.u[j] = *(const unsigned int*)&Vt[f * 16 + ln16][koffB(j, hl)];
            acc[f] = __builtin_amdgcn_wmma_f32_16x16x32_bf16(
                false, pa.v, false, vf.v, (short)0, acc[f], false, false);
        }
    }

    l += __shfl_xor(l, 16, 32);               // merge key-subset halves
    if (hl == 0) Linv[wave][ln16] = 1.0f / l;
    __syncthreads();

    // ---- normalize + store bf16 (acc is standard C-layout: row = query) ----
    unsigned short* op = o + (size_t)(base + q0) * 1024 + h * 64;
#pragma unroll
    for (int v = 0; v < 8; ++v) {
        const int r = v + hl * 8;
        const float inv = Linv[wave][r];
#pragma unroll
        for (int f = 0; f < 4; ++f)
            op[(size_t)r * 1024 + f * 16 + ln16] = f2bf(acc[f][v] * inv);
    }
}

// ---------------------------------------------------------------------------
extern "C" void kernel_launch(void* const* d_in, const int* in_sizes, int n_in,
                              void* d_out, int out_size, void* d_ws,
                              size_t ws_size, hipStream_t stream) {
    const float* x1     = (const float*)d_in[0];
    const float* x2     = (const float*)d_in[1];
    const float* w_qkv  = (const float*)d_in[2];
    const float* b_qkv  = (const float*)d_in[3];
    const float* w_proj = (const float*)d_in[4];
    const float* b_proj = (const float*)d_in[5];
    const float* ln1_w  = (const float*)d_in[6];
    const float* ln1_b  = (const float*)d_in[7];
    const float* ln2_w  = (const float*)d_in[8];
    const float* ln2_b  = (const float*)d_in[9];
    const float* w_fc1  = (const float*)d_in[10];
    const float* b_fc1  = (const float*)d_in[11];
    const float* w_fc2  = (const float*)d_in[12];
    const float* b_fc2  = (const float*)d_in[13];
    const float* gamma1 = (const float*)d_in[14];
    const float* gamma2 = (const float*)d_in[15];

    char* ws = (char*)d_ws;
    size_t off = 0;
    auto alloc = [&](size_t bytes) {
        void* p = ws + off;
        off += (bytes + 255) & ~(size_t)255;
        return p;
    };
    unsigned short* wt_qkv = (unsigned short*)alloc(3072ull * 1024 * 2);
    unsigned short* wt_prj = (unsigned short*)alloc(1024ull * 1024 * 2);
    unsigned short* wt_fc1 = (unsigned short*)alloc(4096ull * 1024 * 2);
    unsigned short* wt_fc2 = (unsigned short*)alloc(1024ull * 4096 * 2);
    unsigned short* lnbuf  = (unsigned short*)alloc(8192ull * 1024 * 2);
    unsigned short* qkvb   = (unsigned short*)alloc(8192ull * 3072 * 2);
    unsigned short* attnb  = (unsigned short*)alloc(8192ull * 1024 * 2);
    float*          xmid   = (float*)alloc(8192ull * 1024 * 4);
    unsigned short* hbuf   = (unsigned short*)alloc(8192ull * 4096 * 2);

    // One-time (per launch) weight convert+transpose to bf16.
    wconv_t<<<(1024u * 3072u) / 256u, 256, 0, stream>>>(w_qkv, wt_qkv, 1024, 3072);
    wconv_t<<<(1024u * 1024u) / 256u, 256, 0, stream>>>(w_proj, wt_prj, 1024, 1024);
    wconv_t<<<(1024u * 4096u) / 256u, 256, 0, stream>>>(w_fc1, wt_fc1, 1024, 4096);
    wconv_t<<<(4096u * 1024u) / 256u, 256, 0, stream>>>(w_fc2, wt_fc2, 4096, 1024);

    const float* xs[2] = {x1, x2};
    const int    Ss[2] = {1024, 512};
    float*       ys[2] = {(float*)d_out, (float*)d_out + 8ull * 1024 * 1024};

    for (int ip = 0; ip < 2; ++ip) {
        const int S = Ss[ip];
        const int M = 8 * S;                    // packed tokens
        const float* x = xs[ip];
        float*       y = ys[ip];

        ln_bf16<<<M, 256, 0, stream>>>(x, ln1_w, ln1_b, lnbuf);

        gemm_bf16<0><<<dim3(M / 128, 3072 / 128), 256, 0, stream>>>(
            lnbuf, wt_qkv, b_qkv, nullptr, nullptr, qkvb, nullptr, M, 3072, 1024);

        attn_fmha<<<dim3(8 * 16, S / 64), 128, 0, stream>>>(qkvb, attnb, S);

        gemm_bf16<2><<<dim3(M / 128, 1024 / 128), 256, 0, stream>>>(
            attnb, wt_prj, b_proj, x, gamma1, nullptr, xmid, M, 1024, 1024);

        ln_bf16<<<M, 256, 0, stream>>>(xmid, ln2_w, ln2_b, lnbuf);

        gemm_bf16<1><<<dim3(M / 128, 4096 / 128), 256, 0, stream>>>(
            lnbuf, wt_fc1, b_fc1, nullptr, nullptr, hbuf, nullptr, M, 4096, 1024);

        gemm_bf16<2><<<dim3(M / 128, 1024 / 128), 256, 0, stream>>>(
            hbuf, wt_fc2, b_fc2, xmid, gamma2, nullptr, y, M, 1024, 4096);
    }
}
